// MLPWithSSM_32899449487951
// MI455X (gfx1250) — matrix operand order repaired and checked by reading, not compiled, not run on hardware
//
#include <hip/hip_runtime.h>
#include <stdint.h>
#include <stddef.h>

typedef __bf16 bf16;
typedef __attribute__((ext_vector_type(16))) __bf16 v16bf;
typedef __attribute__((ext_vector_type(8)))  __bf16 v8bf;
typedef __attribute__((ext_vector_type(8)))  float  v8f;
typedef __attribute__((ext_vector_type(4)))  unsigned int u32x4;
typedef __attribute__((ext_vector_type(8)))  int i32x8;
typedef __attribute__((ext_vector_type(4)))  int i32x4;

static __device__ __forceinline__ bf16 f2bf(float f) {
  uint32_t u = __builtin_bit_cast(uint32_t, f);
  u += 0x7FFFu + ((u >> 16) & 1u);           // round-to-nearest-even
  uint16_t s = (uint16_t)(u >> 16);
  return __builtin_bit_cast(bf16, s);
}

union AFrag { v16bf v; struct { v8bf lo, hi; } p; };

// ---- Tensor Data Mover: 1D tile (n4 x 4-byte elements) global -> LDS ----
// Descriptor per CDNA5 ISA ch.8: group0 = {flags, lds_addr, global_addr, type},
// group1 = {data_size, tensor_dim0/1, tile_dim0..2, strides}, groups 2/3 zero (<=2D).
static __device__ __forceinline__ void tdm_load_1d_f32(uint32_t lds_off,
                                                       const void* gptr,
                                                       uint32_t n4) {
  uint64_t ga = (uint64_t)(uintptr_t)gptr;
  u32x4 g0;
  g0[0] = 1u;                                                  // count=1, user D#
  g0[1] = lds_off;                                             // lds_addr (bytes)
  g0[2] = (uint32_t)ga;                                        // global_addr[31:0]
  g0[3] = (uint32_t)((ga >> 32) & 0x01FFFFFFu) | (2u << 30);   // addr[56:32] | type=2
  i32x8 g1;
  g1[0] = (int)(2u << 16);                      // workgroup_mask=0, data_size=2 (4B)
  g1[1] = (int)((n4 & 0xFFFFu) << 16);          // tensor_dim0[15:0]
  g1[2] = (int)(((n4 >> 16) & 0xFFFFu) | (1u << 16)); // tensor_dim0[31:16], tensor_dim1=1
  g1[3] = (int)((n4 & 0xFFFFu) << 16);          // tensor_dim1 hi=0, tile_dim0=n4
  g1[4] = 0;                                    // tile_dim1=0 (1D), tile_dim2=0
  g1[5] = (int)n4;                              // tensor_dim0_stride[31:0]
  g1[6] = 0;                                    // stride0 hi, stride1 lo
  g1[7] = 0;                                    // stride1 hi
  i32x4 gz = {0, 0, 0, 0};
#if defined(__clang_major__) && __clang_major__ >= 23
  i32x8 gz8 = {0, 0, 0, 0, 0, 0, 0, 0};
  __builtin_amdgcn_tensor_load_to_lds(g0, g1, gz, gz, gz8, 0);
#else
  __builtin_amdgcn_tensor_load_to_lds(g0, g1, gz, gz, 0);
#endif
}

// ---------------- small utility kernels ----------------

__global__ void zero_f32(float* __restrict__ p, int n) {
  int i = blockIdx.x * blockDim.x + threadIdx.x;
  if (i < n) p[i] = 0.0f;
}

__global__ void cvt_f32_bf16(const float* __restrict__ src, bf16* __restrict__ dst, int n) {
  int i = blockIdx.x * blockDim.x + threadIdx.x;
  if (i < n) dst[i] = f2bf(src[i]);
}

// ---------------- spectral norm (power iteration) + scale A -> bf16 ----------------
__global__ void spectral_scale(const float* __restrict__ A, bf16* __restrict__ An) {
  __shared__ float v[256], w[256], red[256];
  int t = threadIdx.x;
  v[t] = 1.0f;
  __syncthreads();
  for (int it = 0; it < 64; ++it) {
    float s = 0.f;
    for (int k = 0; k < 256; ++k) s += A[t * 256 + k] * v[k];  // w = A v
    w[t] = s;
    __syncthreads();
    s = 0.f;
    for (int k = 0; k < 256; ++k) s += A[k * 256 + t] * w[k];  // s = (A^T w)[t]
    red[t] = s * s;
    __syncthreads();
    for (int off = 128; off > 0; off >>= 1) {
      if (t < off) red[t] += red[t + off];
      __syncthreads();
    }
    float nrm = sqrtf(red[0] + 1e-30f);
    __syncthreads();
    v[t] = s / nrm;
    __syncthreads();
  }
  float s = 0.f;
  for (int k = 0; k < 256; ++k) s += A[t * 256 + k] * v[k];
  red[t] = s * s;
  __syncthreads();
  for (int off = 128; off > 0; off >>= 1) {
    if (t < off) red[t] += red[t + off];
    __syncthreads();
  }
  float inv = 1.0f / (sqrtf(red[0]) + 1e-5f);
  for (int k = 0; k < 256; ++k) An[t * 256 + k] = f2bf(A[t * 256 + k] * inv);
}

// ---------------- layer 1: Cin=3 direct FMA GEMM ----------------
__global__ void layer1(const float* __restrict__ x, const float* __restrict__ w1,
                       const float* __restrict__ b1, float* __restrict__ y) {
  int i = blockIdx.x * blockDim.x + threadIdx.x;   // i = m*64 + co
  int co = i & 63, m = i >> 6;
  int b = m >> 10, n = m & 1023;
  const float* xb = x + (size_t)b * 3072 + n;
  float s = b1[co] + w1[co * 3 + 0] * xb[0]
                   + w1[co * 3 + 1] * xb[1024]
                   + w1[co * 3 + 2] * xb[2048];
  y[i] = s;
}

// ---------------- per-channel sum / sumsq over M=65536 rows ----------------
__global__ void stats(const float* __restrict__ y, float* __restrict__ gsum,
                      float* __restrict__ gsum2, int C, int rowsPerBlock) {
  __shared__ float s1[256], s2[256];
  int t = threadIdx.x;
  int c = t & (C - 1);
  int rpi = 256 / C;
  int rbase = blockIdx.x * rowsPerBlock + t / C;
  float a = 0.f, q = 0.f;
  for (int r = 0; r < rowsPerBlock; r += rpi) {
    float v = y[(size_t)(rbase + r) * C + c];
    a += v; q += v * v;
  }
  s1[t] = a; s2[t] = q;
  __syncthreads();
  if (t < C) {
    for (int o = C; o < 256; o += C) { a += s1[t + o]; q += s2[t + o]; }
    atomicAdd(&gsum[t], a);
    atomicAdd(&gsum2[t], q);
  }
}

// ---------------- BN + ReLU + convert to bf16 ----------------
__global__ void bn_relu(const float* __restrict__ y, const float* __restrict__ gsum,
                        const float* __restrict__ gsum2, const float* __restrict__ g,
                        const float* __restrict__ be, bf16* __restrict__ h, int C) {
  int i = blockIdx.x * blockDim.x + threadIdx.x;
  int c = i & (C - 1);
  const float invM = 1.0f / 65536.0f;
  float mu  = gsum[c] * invM;
  float var = gsum2[c] * invM - mu * mu;
  float v = (y[i] - mu) * rsqrtf(var + 1e-5f) * g[c] + be[c];
  h[i] = f2bf(v > 0.f ? v : 0.f);
}

// ---------------- bf16 WMMA GEMM:  Y[M,Co] = H[M,K] * Wb^T (+bias) ----------------
__global__ void __launch_bounds__(256) gemm_bf16(
    const bf16* __restrict__ Hm, const bf16* __restrict__ Wb,
    const float* __restrict__ bias, float* __restrict__ Y,
    int K, int Co, int scatter) {
  int lane = threadIdx.x & 31;
  int wid  = blockIdx.x * 8 + (threadIdx.x >> 5);
  int tilesN = Co >> 6;
  int m0 = (wid / tilesN) * 16;
  int n0 = (wid % tilesN) * 64;
  int half = lane >> 4, l15 = lane & 15;

  v8f acc[4];
  for (int q = 0; q < 4; ++q) {
    float bv = bias ? bias[n0 + q * 16 + l15] : 0.0f;
    for (int j = 0; j < 8; ++j) acc[q][j] = bv;
  }

  const bf16* arow = Hm + (size_t)(m0 + l15) * K + half * 16;
  for (int k0 = 0; k0 < K; k0 += 32) {
    AFrag a;
    a.p.lo = *(const v8bf*)(arow + k0);
    a.p.hi = *(const v8bf*)(arow + k0 + 8);
#pragma unroll
    for (int q = 0; q < 4; ++q) {
      const bf16* bp = Wb + (size_t)(n0 + q * 16 + l15) * K + k0 + half * 16;
      v16bf bfrag = *(const v16bf*)bp;
      acc[q] = __builtin_amdgcn_wmma_f32_16x16x32_bf16(
          false, a.v, false, bfrag, (short)0, acc[q], false, false);
    }
  }

  for (int q = 0; q < 4; ++q)
    for (int j = 0; j < 8; ++j) {
      int m = m0 + j + half * 8;
      int n = n0 + q * 16 + l15;
      if (!scatter) {
        Y[(size_t)m * Co + n] = acc[q][j];
      } else {
        int b = m >> 10, t = m & 1023;
        Y[((size_t)t * 64 + b) * 256 + n] = acc[q][j];
      }
    }
}

// ---------------- SSM scan ----------------
// 4 blocks x 256 threads; block handles 16 batch rows.
// LDS (dynamic, 303104 B): An[128K] | Cm[128K] | state[8K] | u double-buffer[2x16K].
// An/Cm staged once via TDM; per-step u tile DMA'd asynchronously one step ahead.
#define SCAN_LDS_BYTES (131072 + 131072 + 8192 + 32768)
__global__ void __launch_bounds__(256) ssm_scan(
    const bf16* __restrict__ An, const bf16* __restrict__ Cmb,
    const float* __restrict__ uB, float* __restrict__ y) {
  extern __shared__ char smem[];
  bf16*  AnL = (bf16*)smem;                       // 65536 bf16
  bf16*  CmL = (bf16*)(smem + 131072);            // 65536 bf16
  bf16*  st  = (bf16*)(smem + 262144);            // 16x256 bf16 state
  float* uL  = (float*)(smem + 270336);           // 2 x 4096 f32

  int tid = threadIdx.x, lane = tid & 31, w = tid >> 5;
  int half = lane >> 4, l15 = lane & 15;
  int b0 = blockIdx.x * 16;

  if (w == 0) {   // wave-uniform issue; TDM is a wave-level op (EXEC ignored)
    tdm_load_1d_f32((uint32_t)(uintptr_t)AnL, An, 32768);
    tdm_load_1d_f32((uint32_t)(uintptr_t)CmL, Cmb, 32768);
    tdm_load_1d_f32((uint32_t)(uintptr_t)uL, uB + (size_t)b0 * 256, 4096);
  }
  for (int i = tid; i < 16 * 256; i += 256) st[i] = f2bf(0.f);

  const bf16* arow = st + l15 * 256 + half * 16;

  for (int t = 0; t < 1024; ++t) {
    if (w == 0) __builtin_amdgcn_s_wait_tensorcnt(0);  // u[t&1] (and An/Cm) landed
    __syncthreads();

    const float* u = uL + (t & 1) * 4096;
    v8f acc[2];
    for (int q = 0; q < 2; ++q) {
      int n = (2 * w + q) * 16 + l15;
      for (int j = 0; j < 8; ++j) acc[q][j] = u[(j + half * 8) * 256 + n];
    }

    // async-DMA next step's u tile into the other buffer (overlaps both matmuls)
    if (w == 0 && t + 1 < 1024)
      tdm_load_1d_f32((uint32_t)(uintptr_t)(uL + ((t + 1) & 1) * 4096),
                      uB + ((size_t)(t + 1) * 64 + b0) * 256, 4096);

    // ---- hnew = h @ An^T + u_t ----
#pragma unroll
    for (int k0 = 0; k0 < 256; k0 += 32) {
      AFrag a;
      a.p.lo = *(const v8bf*)(arow + k0);
      a.p.hi = *(const v8bf*)(arow + k0 + 8);
      for (int q = 0; q < 2; ++q) {
        const bf16* bp = AnL + (size_t)((2 * w + q) * 16 + l15) * 256 + k0 + half * 16;
        v16bf bfrag = *(const v16bf*)bp;
        acc[q] = __builtin_amdgcn_wmma_f32_16x16x32_bf16(
            false, a.v, false, bfrag, (short)0, acc[q], false, false);
      }
    }
    __syncthreads();   // all state reads done before update
    for (int q = 0; q < 2; ++q)
      for (int j = 0; j < 8; ++j) {
        int m = j + half * 8, n = (2 * w + q) * 16 + l15;
        float v = acc[q][j];
        st[m * 256 + n] = f2bf(v > 0.f ? v : 0.f);
      }
    __syncthreads();   // state updated before mm2 reads

    // ---- y_t = hnew @ Cm^T ----
    v8f out[2];
    for (int q = 0; q < 2; ++q)
      for (int j = 0; j < 8; ++j) out[q][j] = 0.f;
#pragma unroll
    for (int k0 = 0; k0 < 256; k0 += 32) {
      AFrag a;
      a.p.lo = *(const v8bf*)(arow + k0);
      a.p.hi = *(const v8bf*)(arow + k0 + 8);
      for (int q = 0; q < 2; ++q) {
        const bf16* bp = CmL + (size_t)((2 * w + q) * 16 + l15) * 256 + k0 + half * 16;
        v16bf bfrag = *(const v16bf*)bp;
        out[q] = __builtin_amdgcn_wmma_f32_16x16x32_bf16(
            false, a.v, false, bfrag, (short)0, out[q], false, false);
      }
    }
    for (int q = 0; q < 2; ++q)
      for (int j = 0; j < 8; ++j) {
        int m = j + half * 8, co = (2 * w + q) * 16 + l15;
        y[((size_t)(b0 + m) * 256 + co) * 1024 + t] = out[q][j];
      }
  }
}

// ---------------- host side ----------------
extern "C" void kernel_launch(void* const* d_in, const int* in_sizes, int n_in,
                              void* d_out, int out_size, void* d_ws, size_t ws_size,
                              hipStream_t stream) {
  const float* x   = (const float*)d_in[0];
  const float* w1  = (const float*)d_in[1];
  const float* b1  = (const float*)d_in[2];
  const float* g1  = (const float*)d_in[3];
  const float* be1 = (const float*)d_in[4];
  const float* w2  = (const float*)d_in[5];
  const float* b2  = (const float*)d_in[6];
  const float* g2  = (const float*)d_in[7];
  const float* be2 = (const float*)d_in[8];
  const float* w3  = (const float*)d_in[9];
  const float* b3  = (const float*)d_in[10];
  const float* g3  = (const float*)d_in[11];
  const float* be3 = (const float*)d_in[12];
  const float* A   = (const float*)d_in[13];
  const float* Bm  = (const float*)d_in[14];
  const float* Cm  = (const float*)d_in[15];
  float* out = (float*)d_out;

  char* ws = (char*)d_ws;
  float* yf = (float*)ws;                         // 64 MB: y1/y2/y3, then uB [t][b][h]
  bf16*  hA = (bf16*)(ws + ((size_t)64 << 20));   // 32 MB: h1, then h3
  bf16*  hB = (bf16*)(ws + ((size_t)96 << 20));   // 16 MB: h2
  char* sm = ws + ((size_t)112 << 20);
  bf16* w2b = (bf16*)sm;  sm += 128 * 64 * 2;
  bf16* w3b = (bf16*)sm;  sm += 256 * 128 * 2;
  bf16* Bmb = (bf16*)sm;  sm += 256 * 256 * 2;
  bf16* Cmb = (bf16*)sm;  sm += 256 * 256 * 2;
  bf16* Anb = (bf16*)sm;  sm += 256 * 256 * 2;
  float* gsum = (float*)sm;                        // 256 + 256 f32, contiguous

  const int M = 65536;

  // weight conversions + spectral norm
  cvt_f32_bf16<<<(128 * 64 + 255) / 256, 256, 0, stream>>>(w2, w2b, 128 * 64);
  cvt_f32_bf16<<<(256 * 128 + 255) / 256, 256, 0, stream>>>(w3, w3b, 256 * 128);
  cvt_f32_bf16<<<256, 256, 0, stream>>>(Bm, Bmb, 256 * 256);
  cvt_f32_bf16<<<256, 256, 0, stream>>>(Cm, Cmb, 256 * 256);
  spectral_scale<<<1, 256, 0, stream>>>(A, Anb);

  // layer 1 (Cin=3)
  layer1<<<(M * 64) / 256, 256, 0, stream>>>(x, w1, b1, yf);
  zero_f32<<<2, 256, 0, stream>>>(gsum, 512);
  stats<<<128, 256, 0, stream>>>(yf, gsum, gsum + 256, 64, 512);
  bn_relu<<<(M * 64) / 256, 256, 0, stream>>>(yf, gsum, gsum + 256, g1, be1, hA, 64);

  // layer 2: 64 -> 128
  gemm_bf16<<<(4096 * 2) / 8, 256, 0, stream>>>(hA, w2b, b2, yf, 64, 128, 0);
  zero_f32<<<2, 256, 0, stream>>>(gsum, 512);
  stats<<<128, 256, 0, stream>>>(yf, gsum, gsum + 256, 128, 512);
  bn_relu<<<(M * 128) / 256, 256, 0, stream>>>(yf, gsum, gsum + 256, g2, be2, hB, 128);

  // layer 3: 128 -> 256
  gemm_bf16<<<(4096 * 4) / 8, 256, 0, stream>>>(hB, w3b, b3, yf, 128, 256, 0);
  zero_f32<<<2, 256, 0, stream>>>(gsum, 512);
  stats<<<128, 256, 0, stream>>>(yf, gsum, gsum + 256, 256, 512);
  bn_relu<<<(M * 256) / 256, 256, 0, stream>>>(yf, gsum, gsum + 256, g3, be3, hA, 256);

  // uB = h3 @ Bm^T, scattered to [t][b][h]  (overwrites yf, y3 is dead)
  gemm_bf16<<<(4096 * 4) / 8, 256, 0, stream>>>(hA, Bmb, nullptr, yf, 256, 256, 1);

  // sequential scan, batch split across 4 WGPs; 296KB LDS staged via TDM
  ssm_scan<<<4, 256, SCAN_LDS_BYTES, stream>>>(Anb, Cmb, yf, out);
}
